// TransolverAttention_24970939859708
// MI455X (gfx1250) — matrix operand order repaired and checked by reading, not compiled, not run on hardware
//
#include <hip/hip_runtime.h>

// ---------------------------------------------------------------------------
// MI455X (gfx1250, wave32) Transolver attention, bf16 WMMA pipeline
// + async global<->LDS staging (ASYNCcnt) where available.
// ---------------------------------------------------------------------------

typedef __bf16 bf16_t;
typedef __attribute__((ext_vector_type(16))) __bf16 v16bf;
typedef __attribute__((ext_vector_type(8)))  __bf16 v8bf;
typedef __attribute__((ext_vector_type(8)))  float  v8f;

constexpr int kH   = 8;
constexpr int kDH  = 64;
constexpr int kM   = 32;
constexpr int kDIN = 512;
constexpr int kINNER = 512;
constexpr int kB   = 4;
constexpr int kN   = 16384;

#define XS_STRIDE 528   // 512 + 16 pad; 1056B row: 32B-aligned, bank-staggered

// --- gfx1250 async global<->LDS path (ASYNCcnt), guarded by __has_builtin ---
#if defined(__AMDGCN__) && __has_builtin(__builtin_amdgcn_global_load_async_to_lds_b128) && \
    __has_builtin(__builtin_amdgcn_global_store_async_from_lds_b128) && \
    __has_builtin(__builtin_amdgcn_s_wait_asynccnt)
#define HAVE_ASYNC_LDS 1
#endif

#ifdef HAVE_ASYNC_LDS
// exact builtin pointee type: GCC-vector int4 (b128 payload)
typedef int v4i_vs __attribute__((vector_size(16)));
typedef __attribute__((address_space(1))) v4i_vs* gptr128_t;   // global (AS1)
typedef __attribute__((address_space(3))) v4i_vs* lptr128_t;   // LDS (AS3)
#endif

// per-lane 16B async copy global -> LDS (falls back to sync vector copy)
__device__ __forceinline__ void async_load_b128(const void* g, void* l) {
#ifdef HAVE_ASYNC_LDS
  __builtin_amdgcn_global_load_async_to_lds_b128((gptr128_t)(g), (lptr128_t)(l), 0, 0);
#else
  *(v8bf*)l = *(const v8bf*)g;
#endif
}

// per-lane 16B async copy LDS -> global
__device__ __forceinline__ void async_store_b128(void* g, const void* l) {
#ifdef HAVE_ASYNC_LDS
  __builtin_amdgcn_global_store_async_from_lds_b128((gptr128_t)(g), (lptr128_t)(l), 0, 0);
#else
  *(v8bf*)g = *(const v8bf*)l;
#endif
}

__device__ __forceinline__ void wait_async() {
#ifdef HAVE_ASYNC_LDS
  __builtin_amdgcn_s_wait_asynccnt(0);
#endif
}

// --- WMMA fragment helpers (CDNA5 16x16x32 bf16 layouts, ISA 7.12.2) -------
// A (16x32, 16-bit): lane L (0..15) row=L, holds K 0..7 & 16..23;
//                    lane L (16..31) row=L-16, holds K 8..15 & 24..31.
__device__ __forceinline__ v16bf load_a_frag(const bf16_t* base, int rowStride, int lane) {
  const bf16_t* p = base + (lane & 15) * rowStride + ((lane & 16) ? 8 : 0);
  v8bf lo = *(const v8bf*)p;
  v8bf hi = *(const v8bf*)(p + 16);
  v16bf f;
#pragma unroll
  for (int i = 0; i < 8; ++i) { f[i] = lo[i]; f[i + 8] = hi[i]; }
  return f;
}

// B (32x16, 16-bit): lane L holds column (L&15); lanes 0..15 K=0..7 & 16..23,
// lanes 16..31 K=8..15 & 24..31 (same as A). B[k][n] == W[n][k] row-major read.
__device__ __forceinline__ v16bf load_b_frag(const bf16_t* base, long rowStride, int lane) {
  const bf16_t* p = base + (long)(lane & 15) * rowStride + ((lane & 16) ? 8 : 0);
  v8bf lo = *(const v8bf*)p;
  v8bf hi = *(const v8bf*)(p + 16);
  v16bf f;
#pragma unroll
  for (int i = 0; i < 8; ++i) { f[i] = lo[i]; f[i + 8] = hi[i]; }
  return f;
}

__device__ __forceinline__ v8f wmma_bf16(v16bf a, v16bf b, v8f c) {
  // (neg_a, A, neg_b, B, c_mod, C, reuse_a, reuse_b)
  return __builtin_amdgcn_wmma_f32_16x16x32_bf16(false, a, false, b, (short)0, c, false, false);
}

#define V8F_ZERO {0.f,0.f,0.f,0.f,0.f,0.f,0.f,0.f}

// ---------------------------------------------------------------------------
// f32 -> bf16 conversion (grid-stride)
// ---------------------------------------------------------------------------
__global__ __launch_bounds__(256) void k_cvt_bf16(const float* __restrict__ s,
                                                  bf16_t* __restrict__ d, long n) {
  long i = (long)blockIdx.x * blockDim.x + threadIdx.x;
  long stride = (long)gridDim.x * blockDim.x;
  for (; i < n; i += stride) d[i] = (bf16_t)s[i];
}

// ---------------------------------------------------------------------------
// Wcomb[h] = Wslice @ Wx[h]  ([32 x 512] bf16 per head), bcomb = Wslice@bx + bslice
// ---------------------------------------------------------------------------
__global__ __launch_bounds__(128) void k_wcomb(const float* __restrict__ Wx,
                                               const float* __restrict__ bx,
                                               const float* __restrict__ Wslice,
                                               const float* __restrict__ bslice,
                                               bf16_t* __restrict__ WcombB,
                                               float* __restrict__ bcomb) {
  int hm = blockIdx.x;            // 0..255
  int h = hm >> 5, m = hm & 31;
  int tid = threadIdx.x;
  for (int k = tid; k < kDIN; k += 128) {
    float s = 0.f;
    for (int d = 0; d < kDH; ++d)
      s += Wslice[m * kDH + d] * Wx[(h * kDH + d) * kDIN + k];
    WcombB[(h * kM + m) * kDIN + k] = (bf16_t)s;
  }
  if (tid == 0) {
    float s = bslice[m];
    for (int d = 0; d < kDH; ++d) s += Wslice[m * kDH + d] * bx[h * kDH + d];
    bcomb[h * kM + m] = s;
  }
}

// ---------------------------------------------------------------------------
// P1: fused projections + slice softmax.  1 block = 64 tokens, 8 waves.
//   - x tile (64x512 bf16) staged in LDS via async-LDS loads
//   - per head: fx_mid (16 WMMA tiles, K=512), logits (8 WMMA tiles vs Wcomb),
//     softmax over M=32, emit sw, swT, fxT (fxT via async LDS->global store)
// ---------------------------------------------------------------------------
__global__ __launch_bounds__(256) void k_p1(const bf16_t* __restrict__ xB,
                                            const bf16_t* __restrict__ WfxB,
                                            const bf16_t* __restrict__ WcombB,
                                            const float* __restrict__ bcomb,
                                            const float* __restrict__ bfx,
                                            const float* __restrict__ temp,
                                            bf16_t* __restrict__ swG,
                                            bf16_t* __restrict__ swTG,
                                            bf16_t* __restrict__ fxTG) {
  __shared__ __align__(32) bf16_t XS[64 * XS_STRIDE];   // 67,584 B
  __shared__ __align__(32) float  LOGS[64 * 36];        //  9,216 B
  __shared__ __align__(32) bf16_t SWS[64 * 40];         //  5,120 B
  __shared__ __align__(32) bf16_t FXT[64 * 80];         // 10,240 B  (FXT[d][t])

  const int tid = threadIdx.x;
  const int wave = tid >> 5, lane = tid & 31;
  const int tile = blockIdx.x;
  const int b = tile >> 8;
  const int n0 = (tile & 255) * 64;

  // stage x tile: 64 rows x 512 bf16 = 64KB, async 16B/lane chunks
  {
    const bf16_t* src = xB + (long)(b * kN + n0) * kDIN;
#pragma unroll
    for (int it = 0; it < 16; ++it) {
      int idx = it * 256 + tid;        // 0..4095
      int row = idx >> 6;              // 64 chunks per row
      int col = (idx & 63) * 8;        // 8 bf16 = 16B per chunk
      async_load_b128(src + (long)row * kDIN + col, &XS[row * XS_STRIDE + col]);
    }
    wait_async();
  }
  __syncthreads();

  for (int h = 0; h < kH; ++h) {
    const long bh = (long)(b * kH + h);

    // ---- phase A: 24 WMMA tile-jobs (16 FX + 8 LOG), 3 per wave ----
    for (int jj = 0; jj < 3; ++jj) {
      int j = wave + jj * 8;
      if (j < 16) {
        int r = j >> 2, c = j & 3;
        const bf16_t* Bb = WfxB + (long)(h * 64 + c * 16) * kDIN;
        v8f acc = V8F_ZERO;
        for (int k = 0; k < kDIN; k += 32) {
          __builtin_prefetch(Bb + (long)(lane & 15) * kDIN + k + 128, 0, 1);
          v16bf a = load_a_frag(&XS[(r * 16) * XS_STRIDE + k], XS_STRIDE, lane);
          v16bf bb = load_b_frag(Bb + k, kDIN, lane);
          acc = wmma_bf16(a, bb, acc);
        }
        int col = lane & 15;
        int rbase = (lane & 16) ? 8 : 0;
        int d = c * 16 + col;
        float bias = bfx[h * 64 + d];
#pragma unroll
        for (int i = 0; i < 8; ++i) {
          int t = r * 16 + rbase + i;
          FXT[d * 80 + t] = (bf16_t)(acc[i] + bias);   // store transposed
        }
      } else {
        int q = j - 16;
        int r = q >> 1, c = q & 1;
        const bf16_t* Bb = WcombB + (long)(h * 32 + c * 16) * kDIN;
        v8f acc = V8F_ZERO;
        for (int k = 0; k < kDIN; k += 32) {
          v16bf a = load_a_frag(&XS[(r * 16) * XS_STRIDE + k], XS_STRIDE, lane);
          v16bf bb = load_b_frag(Bb + k, kDIN, lane);
          acc = wmma_bf16(a, bb, acc);
        }
        int col = lane & 15;
        int rbase = (lane & 16) ? 8 : 0;
        int m = c * 16 + col;
        float bias = bcomb[h * 32 + m];
#pragma unroll
        for (int i = 0; i < 8; ++i)
          LOGS[(r * 16 + rbase + i) * 36 + m] = acc[i] + bias;
      }
    }
    __syncthreads();

    // ---- phase B: softmax over M (threads 0..63 = tokens) + fxT flush ----
    if (tid < 64) {
      int t = tid;
      float invT = 1.0f / temp[h];
      float l[32];
      float mx = -1e30f;
#pragma unroll
      for (int m = 0; m < 32; ++m) { l[m] = LOGS[t * 36 + m] * invT; mx = fmaxf(mx, l[m]); }
      float s = 0.f;
#pragma unroll
      for (int m = 0; m < 32; ++m) { l[m] = __expf(l[m] - mx); s += l[m]; }
      float inv = 1.0f / s;
      bf16_t* gp = swG + (bh * kN + n0 + t) * kM;
#pragma unroll
      for (int v = 0; v < 4; ++v) {
        v8bf pk;
#pragma unroll
        for (int i = 0; i < 8; ++i) {
          bf16_t w = (bf16_t)(l[v * 8 + i] * inv);
          SWS[t * 40 + v * 8 + i] = w;
          pk[i] = w;
        }
        *(v8bf*)(gp + v * 8) = pk;
      }
    }
    { // flush fxT tile (8KB) LDS -> global, async 16B chunks
#pragma unroll
      for (int it = 0; it < 2; ++it) {
        int idx = it * 256 + tid;      // 0..511
        int d = idx >> 3, sub = idx & 7;
        int t0 = sub * 8;
        async_store_b128(fxTG + (bh * kDH + d) * (long)kN + n0 + t0,
                         &FXT[d * 80 + t0]);
      }
    }
    __syncthreads();

    // ---- phase C: swT flush (transposed gather from LDS, 16B stores) ----
    {
      int m = tid >> 3, sub = tid & 7;
      int t0 = sub * 8;
      v8bf v;
#pragma unroll
      for (int i = 0; i < 8; ++i) v[i] = SWS[(t0 + i) * 40 + m];
      bf16_t* gp = swTG + (bh * kM + m) * (long)kN + n0 + t0;
      *(v8bf*)gp = v;
    }
    wait_async();           // FXT consumed before next head overwrites it
    __syncthreads();
  }
}

// ---------------------------------------------------------------------------
// P2a: slice_token = swT @ fxT^T over N (deterministic fixed-order WMMA chain)
//      + slice_norm.  1 block = one (b,h); 8 waves = 8 C-tiles of [32 x 64].
// ---------------------------------------------------------------------------
__global__ __launch_bounds__(256) void k_p2a(const bf16_t* __restrict__ swT,
                                             const bf16_t* __restrict__ fxT,
                                             float* __restrict__ stG,
                                             float* __restrict__ normG) {
  __shared__ float NP[8][32];
  const int tid = threadIdx.x, wave = tid >> 5, lane = tid & 31;
  const long bh = blockIdx.x;
  const int rm = wave >> 2, cd = wave & 3;

  const bf16_t* aBase = swT + (bh * kM + rm * 16) * (long)kN;
  const bf16_t* bBase = fxT + (bh * kDH + cd * 16) * (long)kN;
  v8f acc = V8F_ZERO;
  for (int n = 0; n < kN; n += 32) {
    v16bf a = load_a_frag(aBase + n, kN, lane);
    v16bf bb = load_b_frag(bBase + n, kN, lane);
    acc = wmma_bf16(a, bb, acc);
  }
  {
    int col = lane & 15, rbase = (lane & 16) ? 8 : 0;
    float* gp = stG + bh * (kM * kDH);
#pragma unroll
    for (int i = 0; i < 8; ++i)
      gp[(rm * 16 + rbase + i) * kDH + cd * 16 + col] = acc[i];
  }
  // norm[m] = sum_n sw[n][m] : 8 chunks x 32 m
  {
    int m = tid & 31, chunk = tid >> 5;
    const bf16_t* p = swT + (bh * kM + m) * (long)kN + chunk * (kN / 8);
    float s = 0.f;
    for (int i = 0; i < kN / 8; i += 8) {
      v8bf v = *(const v8bf*)(p + i);
#pragma unroll
      for (int jv = 0; jv < 8; ++jv) s += (float)v[jv];
    }
    NP[chunk][m] = s;
  }
  __syncthreads();
  if (tid < 32) {
    float s = 0.f;
#pragma unroll
    for (int c2 = 0; c2 < 8; ++c2) s += NP[c2][tid];
    normG[bh * kM + tid] = s;
  }
}

// ---------------------------------------------------------------------------
// P2b: cluster attention per (b,h): q/k/v, scores softmax, out = P@V.
//      Writes out_slice TRANSPOSED (osT[d][m]) for P3's B-fragments.
// ---------------------------------------------------------------------------
__global__ __launch_bounds__(256) void k_p2b(const float* __restrict__ stG,
                                             const float* __restrict__ normG,
                                             const bf16_t* __restrict__ WqB,
                                             const bf16_t* __restrict__ WkB,
                                             const bf16_t* __restrict__ WvB,
                                             bf16_t* __restrict__ osTG) {
  __shared__ __align__(32) bf16_t STb[32 * 80];
  __shared__ __align__(32) bf16_t Qs[32 * 80];
  __shared__ __align__(32) bf16_t Ks[32 * 80];
  __shared__ __align__(32) bf16_t VT[64 * 48];
  __shared__ __align__(32) float  SS[32 * 36];
  __shared__ __align__(32) bf16_t Pb[32 * 48];

  const int tid = threadIdx.x, wave = tid >> 5, lane = tid & 31;
  const long bh = blockIdx.x;

  { // load slice_token, normalize, convert bf16
    const float* sp = stG + bh * (kM * kDH);
#pragma unroll
    for (int it = 0; it < 8; ++it) {
      int idx = it * 256 + tid;           // 0..2047
      int m = idx >> 6, d = idx & 63;
      float nv = normG[bh * kM + m] + 1e-5f;
      STb[m * 80 + d] = (bf16_t)(sp[idx] / nv);
    }
  }
  __syncthreads();

  // Q/K/V: 24 tile jobs (3 per wave), K=64
#pragma unroll
  for (int jj = 0; jj < 3; ++jj) {
    int j = wave + jj * 8;
    int which = j >> 3;                   // 0=Q 1=K 2=V
    int t = j & 7;
    int rm = t >> 2, cd = t & 3;
    const bf16_t* W = (which == 0) ? WqB : (which == 1) ? WkB : WvB;
    const bf16_t* Bb = W + (cd * 16) * kDH;
    v8f acc = V8F_ZERO;
#pragma unroll
    for (int k = 0; k < kDH; k += 32) {
      v16bf a = load_a_frag(&STb[(rm * 16) * 80 + k], 80, lane);
      v16bf bb = load_b_frag(Bb + k, kDH, lane);
      acc = wmma_bf16(a, bb, acc);
    }
    int col = lane & 15, rbase = (lane & 16) ? 8 : 0;
#pragma unroll
    for (int i = 0; i < 8; ++i) {
      int m = rm * 16 + rbase + i, d = cd * 16 + col;
      bf16_t v = (bf16_t)acc[i];
      if (which == 0)      Qs[m * 80 + d] = v;
      else if (which == 1) Ks[m * 80 + d] = v;
      else                 VT[d * 48 + m] = v;      // V transposed
    }
  }
  __syncthreads();

  // scores S = Q @ K^T (4 tiles, waves 0..3), K=64
  if (wave < 4) {
    int rm = wave >> 1, cj = wave & 1;
    v8f acc = V8F_ZERO;
#pragma unroll
    for (int k = 0; k < kDH; k += 32) {
      v16bf a = load_a_frag(&Qs[(rm * 16) * 80 + k], 80, lane);
      v16bf bb = load_b_frag(&Ks[(cj * 16) * 80 + k], 80, lane);
      acc = wmma_bf16(a, bb, acc);
    }
    int col = lane & 15, rbase = (lane & 16) ? 8 : 0;
#pragma unroll
    for (int i = 0; i < 8; ++i)
      SS[(rm * 16 + rbase + i) * 36 + cj * 16 + col] = acc[i];
  }
  __syncthreads();

  if (tid < 32) {                          // softmax rows, scale = DH^-0.5
    const float scale = 0.125f;
    float l[32];
    float mx = -1e30f;
#pragma unroll
    for (int jm = 0; jm < 32; ++jm) { l[jm] = SS[tid * 36 + jm] * scale; mx = fmaxf(mx, l[jm]); }
    float s = 0.f;
#pragma unroll
    for (int jm = 0; jm < 32; ++jm) { l[jm] = __expf(l[jm] - mx); s += l[jm]; }
    float inv = 1.f / s;
#pragma unroll
    for (int jm = 0; jm < 32; ++jm) Pb[tid * 48 + jm] = (bf16_t)(l[jm] * inv);
  }
  __syncthreads();

  // out = P @ V (8 tiles, K=32 single WMMA), store transposed to global
  {
    int rm = wave >> 2, cd = wave & 3;
    v16bf a = load_a_frag(&Pb[(rm * 16) * 48], 48, lane);
    v16bf bb = load_b_frag(&VT[(cd * 16) * 48], 48, lane);
    v8f acc = V8F_ZERO;
    acc = wmma_bf16(a, bb, acc);
    int col = lane & 15, rbase = (lane & 16) ? 8 : 0;
    bf16_t* gp = osTG + bh * (kDH * kM);
#pragma unroll
    for (int i = 0; i < 8; ++i) {
      int m = rm * 16 + rbase + i, d = cd * 16 + col;
      gp[d * kM + m] = (bf16_t)acc[i];
    }
  }
}

// ---------------------------------------------------------------------------
// P3: fused de-slice + output GEMM.  1 block = 64 tokens.
//   per head: out_x block [64 x 64] = sw[64x32] @ out_slice[32x64] (K=32 WMMA)
//   then OUT[64x512] = out_x @ Wout^T + bout (K=512 WMMA chain), f32 to d_out.
//   sw / out_sliceT tiles staged via async-LDS loads.
// ---------------------------------------------------------------------------
__global__ __launch_bounds__(256) void k_p3(const bf16_t* __restrict__ swG,
                                            const bf16_t* __restrict__ osTG,
                                            const bf16_t* __restrict__ WoutB,
                                            const float* __restrict__ bout,
                                            float* __restrict__ out) {
  __shared__ __align__(32) bf16_t SWH[64 * 40];          //  5,120 B
  __shared__ __align__(32) bf16_t OSTH[64 * 48];         //  6,144 B
  __shared__ __align__(32) bf16_t OX[64 * XS_STRIDE];    // 67,584 B

  const int tid = threadIdx.x, wave = tid >> 5, lane = tid & 31;
  const int tile = blockIdx.x;
  const int b = tile >> 8;
  const int n0 = (tile & 255) * 64;

  for (int h = 0; h < kH; ++h) {
    const long bh = (long)(b * kH + h);
    { // async-stage sw tile (64 tok x 32) and out_sliceT (64 d x 32 m)
      int t = tid >> 2, q = tid & 3;
      async_load_b128(swG + (bh * kN + n0 + t) * kM + q * 8, &SWH[t * 40 + q * 8]);
      async_load_b128(osTG + bh * (kDH * kM) + t * kM + q * 8, &OSTH[t * 48 + q * 8]);
      wait_async();
    }
    __syncthreads();
#pragma unroll
    for (int jj = 0; jj < 2; ++jj) {      // 16 tiles, 2 per wave, K=32
      int j = wave * 2 + jj;
      int r = j >> 2, c = j & 3;
      v16bf a = load_a_frag(&SWH[(r * 16) * 40], 40, lane);
      v16bf bb = load_b_frag(&OSTH[(c * 16) * 48], 48, lane);
      v8f acc = V8F_ZERO;
      acc = wmma_bf16(a, bb, acc);
      int col = lane & 15, rbase = (lane & 16) ? 8 : 0;
#pragma unroll
      for (int i = 0; i < 8; ++i)
        OX[(r * 16 + rbase + i) * XS_STRIDE + h * 64 + c * 16 + col] = (bf16_t)acc[i];
    }
    __syncthreads();
  }

  // final GEMM: 128 C-tiles (4 x 32), 16 per wave, K=512
  for (int jj = 0; jj < 16; ++jj) {
    int j = wave * 16 + jj;
    int r = j >> 5, c = j & 31;
    const bf16_t* Bb = WoutB + (long)(c * 16) * kINNER;
    v8f acc = V8F_ZERO;
    for (int k = 0; k < kINNER; k += 32) {
      __builtin_prefetch(Bb + (long)(lane & 15) * kINNER + k + 128, 0, 1);
      v16bf a = load_a_frag(&OX[(r * 16) * XS_STRIDE + k], XS_STRIDE, lane);
      v16bf bb = load_b_frag(Bb + k, kINNER, lane);
      acc = wmma_bf16(a, bb, acc);
    }
    int col = lane & 15, rbase = (lane & 16) ? 8 : 0;
    int dout = c * 16 + col;
    float bias = bout[dout];
    float* gp = out + (long)(b * kN + n0 + r * 16 + rbase) * kDIN;
#pragma unroll
    for (int i = 0; i < 8; ++i)
      gp[(long)i * kDIN + dout] = acc[i] + bias;
  }
}

// ---------------------------------------------------------------------------
extern "C" void kernel_launch(void* const* d_in, const int* in_sizes, int n_in,
                              void* d_out, int out_size, void* d_ws, size_t ws_size,
                              hipStream_t stream) {
  const float* x      = (const float*)d_in[0];
  const float* Wx     = (const float*)d_in[1];
  const float* bx     = (const float*)d_in[2];
  const float* Wfx    = (const float*)d_in[3];
  const float* bfx    = (const float*)d_in[4];
  const float* Wslice = (const float*)d_in[5];
  const float* bslice = (const float*)d_in[6];
  const float* Wq     = (const float*)d_in[7];
  const float* Wk     = (const float*)d_in[8];
  const float* Wv     = (const float*)d_in[9];
  const float* Wout   = (const float*)d_in[10];
  const float* bout   = (const float*)d_in[11];
  const float* temp   = (const float*)d_in[12];
  (void)in_sizes; (void)n_in; (void)out_size; (void)ws_size;

  size_t off = 0;
  char* base = (char*)d_ws;
  auto take = [&](size_t bytes) -> void* {
    void* p = base + off;
    off += (bytes + 255) & ~(size_t)255;
    return p;
  };

  bf16_t* xB     = (bf16_t*)take((size_t)kB * kN * kDIN * 2);        // 64 MB
  bf16_t* WfxB   = (bf16_t*)take((size_t)kINNER * kDIN * 2);
  bf16_t* WcombB = (bf16_t*)take((size_t)kH * kM * kDIN * 2);
  float*  bcomb  = (float*) take((size_t)kH * kM * 4);
  bf16_t* WqB    = (bf16_t*)take((size_t)kDH * kDH * 2);
  bf16_t* WkB    = (bf16_t*)take((size_t)kDH * kDH * 2);
  bf16_t* WvB    = (bf16_t*)take((size_t)kDH * kDH * 2);
  bf16_t* WoutB  = (bf16_t*)take((size_t)kDIN * kINNER * 2);
  bf16_t* swB    = (bf16_t*)take((size_t)kB * kH * kN * kM * 2);     // 32 MB
  bf16_t* swTB   = (bf16_t*)take((size_t)kB * kH * kM * kN * 2);     // 32 MB
  bf16_t* fxTB   = (bf16_t*)take((size_t)kB * kH * kDH * kN * 2);    // 64 MB
  float*  stF    = (float*) take((size_t)kB * kH * kM * kDH * 4);
  float*  normF  = (float*) take((size_t)kB * kH * kM * 4);
  bf16_t* osTB   = (bf16_t*)take((size_t)kB * kH * kDH * kM * 2);

  // conversions
  k_cvt_bf16<<<4096, 256, 0, stream>>>(x,    xB,    (long)kB * kN * kDIN);
  k_cvt_bf16<<<512,  256, 0, stream>>>(Wfx,  WfxB,  (long)kINNER * kDIN);
  k_cvt_bf16<<<16,   256, 0, stream>>>(Wq,   WqB,   (long)kDH * kDH);
  k_cvt_bf16<<<16,   256, 0, stream>>>(Wk,   WkB,   (long)kDH * kDH);
  k_cvt_bf16<<<16,   256, 0, stream>>>(Wv,   WvB,   (long)kDH * kDH);
  k_cvt_bf16<<<512,  256, 0, stream>>>(Wout, WoutB, (long)kDIN * kINNER);

  // folded logits weights
  k_wcomb<<<kH * kM, 128, 0, stream>>>(Wx, bx, Wslice, bslice, WcombB, bcomb);

  // fused projections + slice softmax
  k_p1<<<kB * (kN / 64), 256, 0, stream>>>(xB, WfxB, WcombB, bcomb, bfx, temp,
                                           swB, swTB, fxTB);
  // deterministic slice_token + norm reduction
  k_p2a<<<kB * kH, 256, 0, stream>>>(swTB, fxTB, stF, normF);
  // cluster attention
  k_p2b<<<kB * kH, 256, 0, stream>>>(stF, normF, WqB, WkB, WvB, osTB);
  // de-slice + output GEMM
  k_p3<<<kB * (kN / 64), 256, 0, stream>>>(swB, osTB, WoutB, bout, (float*)d_out);
}
